// SeparateWindowedValuesCoordinatesAttention_68444598829560
// MI455X (gfx1250) — matrix operand order repaired and checked by reading, not compiled, not run on hardware
//
#include <hip/hip_runtime.h>

// ---------------------------------------------------------------------------
// Types for CDNA5 WMMA (wave32): D(16x16,f32) = A(16x32,bf16) x B(32x16,bf16) + C
// ---------------------------------------------------------------------------
typedef __attribute__((ext_vector_type(16))) __bf16       v16bf;
typedef __attribute__((ext_vector_type(8)))  float        v8f;
typedef __attribute__((ext_vector_type(4)))  unsigned int v4u;

union FragU { v16bf bf; v4u u4[2]; };

__device__ __forceinline__ unsigned short f2bf(float f) {
  unsigned int u = __builtin_bit_cast(unsigned int, f);
  u += 0x7fffu + ((u >> 16) & 1u);              // round-to-nearest-even
  return (unsigned short)(u >> 16);
}
__device__ __forceinline__ float bf2f(unsigned short h) {
  unsigned int u = ((unsigned int)h) << 16;
  return __builtin_bit_cast(float, u);
}

__device__ __forceinline__ v16bf ld2(const unsigned short* p0, const unsigned short* p1) {
  FragU f;
  f.u4[0] = *(const v4u*)p0;
  f.u4[1] = *(const v4u*)p1;
  return f.bf;
}

// A fragment 16x32 (MxK) at (m0,k0) from row-major bf16 tile, row stride ld.
__device__ __forceinline__ v16bf fragA(const unsigned short* s, int ld, int m0, int k0) {
  int l = threadIdx.x & 31;
  const unsigned short* b = s + (m0 + (l & 15)) * ld + k0 + ((l >> 4) << 4);
  return ld2(b, b + 8);
}
// B fragment 32x16 (KxN) at (k0,n0); tile stored as [n][k] row-major (K contiguous).
__device__ __forceinline__ v16bf fragB(const unsigned short* s, int ld, int n0, int k0) {
  int l = threadIdx.x & 31;
  const unsigned short* b = s + (n0 + (l & 15)) * ld + k0 + ((l >> 4) << 4);
  return ld2(b, b + 8);
}
// B fragment from pre-packed weights: per (kt,nt) tile, 32 lanes x 16 bf16 contiguous.
__device__ __forceinline__ v16bf fragBg(const unsigned short* W, int Ntiles, int kt, int nt) {
  const unsigned short* b = W + ((((long)(kt * Ntiles + nt)) << 5) + (threadIdx.x & 31)) * 16;
  return ld2(b, b + 8);
}
__device__ __forceinline__ v8f wmma_bf16(v16bf a, v16bf b, v8f c) {
  return __builtin_amdgcn_wmma_f32_16x16x32_bf16(false, a, false, b, (short)0, c, false, false);
}

// ---------------------------------------------------------------------------
// gfx1250 async global->LDS copy (ASYNCcnt-tracked, no VGPR round trip).
// LDS flat addresses use addr[31:0] as the wave-relative LDS byte offset.
// ---------------------------------------------------------------------------
__device__ __forceinline__ void async_ld_b128(const void* gptr, void* lptr) {
  unsigned lds_off = (unsigned)(unsigned long long)lptr;
  asm volatile("global_load_async_to_lds_b128 %0, %1, off"
               :: "v"(lds_off), "v"((unsigned long long)gptr)
               : "memory");
}
__device__ __forceinline__ void wait_async0() {
  asm volatile("s_wait_asynccnt 0x0" ::: "memory");
}

// ---------------------------------------------------------------------------
// Problem constants
// ---------------------------------------------------------------------------
#define NTOK  100352   // 2*224*224
#define NWH   12544    // 2*8*28*28 window-heads

// ---------------------------------------------------------------------------
// fp32 -> bf16 convert (activations)
// ---------------------------------------------------------------------------
__global__ __launch_bounds__(256) void cvt_bf16_kernel(const float* __restrict__ src,
                                                       unsigned short* __restrict__ dst,
                                                       long n) {
  long i = (long)blockIdx.x * 256 + threadIdx.x;
  long stride = (long)gridDim.x * 256;
  for (; i < n; i += stride) dst[i] = f2bf(src[i]);
}

// ---------------------------------------------------------------------------
// Pack weights (two row-major KxN0 / KxN1 halves concatenated along N) into
// WMMA-B fragment order: [(kt*Ntiles+nt)*512 + lane*16 + e]
// ---------------------------------------------------------------------------
__global__ __launch_bounds__(256) void pack_w_kernel(const float* __restrict__ W0, int N0,
                                                     const float* __restrict__ W1, int N1,
                                                     unsigned short* __restrict__ dst,
                                                     int K, int N) {
  long total = (long)K * N;
  long i = (long)blockIdx.x * 256 + threadIdx.x;
  long stride = (long)gridDim.x * 256;
  int Ntiles = N >> 4;
  for (; i < total; i += stride) {
    int e = (int)(i & 15);
    int lane = (int)((i >> 4) & 31);
    long t = i >> 9;
    int nt = (int)(t % Ntiles);
    int kt = (int)(t / Ntiles);
    int k = (kt << 5) + ((lane >> 4) << 4) + e;
    int n = (nt << 4) + (lane & 15);
    float v = (n < N0) ? W0[(long)k * N0 + n] : W1[(long)k * N1 + (n - N0)];
    dst[i] = f2bf(v);
  }
}

// ---------------------------------------------------------------------------
// Fused projection: P = Xb(64xK) @ W(KxN) + bias, rmsnorm over first normN cols,
// interleaved channel split (IL=3: q/k/v, IL=2: qc/kc), scatter into windowed
// layout [b][h][wy][wx][p][e] bf16 (v stored transposed [..][e][p]).
// Folds the -4/-4 roll into the scatter.
// ---------------------------------------------------------------------------
__global__ __launch_bounds__(256)
void proj_kernel(const unsigned short* __restrict__ Xb, int K,
                 const unsigned short* __restrict__ Wp, int N, int normN,
                 const float* __restrict__ bias0, const float* __restrict__ bias1,
                 const float* __restrict__ g, int IL,
                 unsigned short* __restrict__ dq, unsigned short* __restrict__ dk,
                 unsigned short* __restrict__ dv) {
  extern __shared__ unsigned char smemraw[];
  const int ldA = K + 8;                 // bank-rotated stride
  const int ldP = N + 8;
  unsigned short* As = (unsigned short*)smemraw;              // 64 x ldA bf16
  unsigned short* Ps = As + 64 * ldA;                         // 64 x ldP bf16
  float* sumsq = (float*)(Ps + 64 * ldP);                     // 64 floats

  const int mt0 = blockIdx.x << 6;

  // Stage 64xK activation tile via async global->LDS b128 transfers
  const int kv8 = K >> 3;
  for (int i = threadIdx.x; i < (kv8 << 6); i += 256) {
    int row = i / kv8, cv = i - row * kv8;
    async_ld_b128(Xb + (long)(mt0 + row) * K + (cv << 3),
                  As + row * ldA + (cv << 3));
  }
  if (threadIdx.x < 64) sumsq[threadIdx.x] = 0.f;
  wait_async0();
  __syncthreads();

  const int wave = threadIdx.x >> 5, lane = threadIdx.x & 31;
  const int wm = (wave & 3) << 4;        // M sub-tile 0/16/32/48
  const int wn = (wave >> 2) << 5;       // N half 0/32
  const int hi8 = (lane >> 4) << 3;
  const int nl = lane & 15;
  const int Kt = K >> 5, Ntiles = N >> 4;

  for (int nb = 0; nb < N; nb += 64) {
    const int nt0 = (nb + wn) >> 4, nt1 = nt0 + 1;
    v8f acc0 = {}, acc1 = {};
    // software-pipelined B fragments: next kt's loads issue before current WMMA
    v16bf b0 = fragBg(Wp, Ntiles, 0, nt0);
    v16bf b1 = fragBg(Wp, Ntiles, 0, nt1);
    for (int kt = 0; kt < Kt; ++kt) {
      v16bf a = fragA(As, ldA, wm, kt << 5);
      v16bf c0 = b0, c1 = b1;
      if (kt + 1 < Kt) {
        b0 = fragBg(Wp, Ntiles, kt + 1, nt0);
        b1 = fragBg(Wp, Ntiles, kt + 1, nt1);
      }
      acc0 = wmma_bf16(a, c0, acc0);
      acc1 = wmma_bf16(a, c1, acc1);
    }
    for (int f = 0; f < 2; ++f) {
      v8f acc = f ? acc1 : acc0;
      int col = nb + wn + (f << 4) + nl;
      float bb = (col < normN) ? bias0[col] : bias1[col - normN];
      for (int r = 0; r < 8; ++r) {
        int row = wm + hi8 + r;
        float v = acc[r] + bb;
        if (col < normN) atomicAdd(&sumsq[row], v * v);   // ds_add_f32
        Ps[row * ldP + col] = f2bf(v);
      }
    }
  }
  __syncthreads();
  if (threadIdx.x < 64)
    sumsq[threadIdx.x] = rsqrtf(sumsq[threadIdx.x] / (float)normN + 1e-6f);
  __syncthreads();

  // Scatter: channel j -> (d=j/IL, r=j%IL); d = e*8 + head (head fastest);
  // token -> rolled coords -> window (wy,wx), in-window p = iy*8+ix.
  const long tot = (long)N << 6;
  for (long i = threadIdx.x; i < tot; i += 256) {
    int row = (int)(i / N);
    int j = (int)(i - (long)row * N);
    float v = bf2f(Ps[row * ldP + j]);
    if (j < normN) v = v * g[j] * sumsq[row];
    int d = j / IL, r = j - d * IL;
    int e = d >> 3, hh = d & 7;
    int t = mt0 + row;
    int bI = t / 50176;
    int rem = t - bI * 50176;
    int y = rem / 224, x = rem - y * 224;
    int yr = y - 4; if (yr < 0) yr += 224;
    int xr = x - 4; if (xr < 0) xr += 224;
    int wy = yr >> 3, iy = yr & 7, wx = xr >> 3, ix = xr & 7;
    int p = (iy << 3) | ix;
    int base = ((((bI << 3) | hh) * 28 + wy) * 28 + wx) << 6;
    unsigned short o = f2bf(v);
    if (r == 0)      dq[((long)(base + p) << 6) + e] = o;
    else if (r == 1) dk[((long)(base + p) << 6) + e] = o;
    else             dv[((long)(base + e) << 6) + p] = o;   // v transposed [e][p]
  }
}

// ---------------------------------------------------------------------------
// Attention: one block per window-head.
// S = (q kT + qc kcT)/8 + posbias (+ shifted-window masks), softmax, Y = S v.
// Writes Y directly in output channel layout c = head*64 + e at rolled coords.
// ---------------------------------------------------------------------------
__global__ __launch_bounds__(256)
void attn_kernel(const unsigned short* __restrict__ qb, const unsigned short* __restrict__ kb,
                 const unsigned short* __restrict__ qcb, const unsigned short* __restrict__ kcb,
                 const unsigned short* __restrict__ vtb, const float* __restrict__ pos,
                 unsigned short* __restrict__ Yb) {
  extern __shared__ unsigned char smemraw[];
  float* posS = (float*)smemraw;                       // 256 f
  float* Sf   = posS + 256;                            // 64 x 68 f
  unsigned short* qs  = (unsigned short*)(Sf + 64 * 68);
  unsigned short* ks  = qs  + 64 * 72;
  unsigned short* qcs = ks  + 64 * 72;
  unsigned short* kcs = qcs + 64 * 72;
  unsigned short* vts = kcs + 64 * 72;
  unsigned short* ab  = vts + 64 * 72;                 // attn bf16

  const int wi = blockIdx.x;
  const int wx = wi % 28, wy = (wi / 28) % 28;
  const int hh = (wi / 784) & 7, bI = wi / 6272;
  const long gbase = (long)wi << 12;                   // * 4096

  // Stage all five 64x64 bf16 tiles via async global->LDS transfers
  for (int i = threadIdx.x; i < 512; i += 256) {
    int row = i >> 3, cv = i & 7;
    long go = gbase + (row << 6) + (cv << 3);
    int lo = row * 72 + (cv << 3);
    async_ld_b128(qb + go, qs + lo);
    async_ld_b128(kb + go, ks + lo);
    async_ld_b128(qcb + go, qcs + lo);
    async_ld_b128(kcb + go, kcs + lo);
    async_ld_b128(vtb + go, vts + lo);
  }
  if (threadIdx.x < 225) posS[threadIdx.x] = pos[threadIdx.x];
  wait_async0();
  __syncthreads();

  const int wave = threadIdx.x >> 5, lane = threadIdx.x & 31;
  const int wm = (wave & 3) << 4, wn = (wave >> 2) << 5;
  const int hi8 = (lane >> 4) << 3, nl = lane & 15;
  const int n0 = wn, n1 = wn + 16;

  // logits: dual-N accumulators, A fragments shared across both N tiles
  {
    v8f acc0 = {}, acc1 = {};
    for (int kt = 0; kt < 2; ++kt) {
      v16bf aq = fragA(qs, 72, wm, kt << 5);
      v16bf ac = fragA(qcs, 72, wm, kt << 5);
      acc0 = wmma_bf16(aq, fragB(ks, 72, n0, kt << 5), acc0);
      acc1 = wmma_bf16(aq, fragB(ks, 72, n1, kt << 5), acc1);
      acc0 = wmma_bf16(ac, fragB(kcs, 72, n0, kt << 5), acc0);
      acc1 = wmma_bf16(ac, fragB(kcs, 72, n1, kt << 5), acc1);
    }
    for (int f = 0; f < 2; ++f) {
      v8f acc = f ? acc1 : acc0;
      int q = wn + (f << 4) + nl;
      int qyc = q >> 3, qxc = q & 7;
      for (int r = 0; r < 8; ++r) {
        int p = wm + hi8 + r;
        float v = acc[r] * 0.125f;                      // / sqrt(64)
        v += posS[(qyc - (p >> 3) + 7) * 15 + (qxc - (p & 7) + 7)];
        if (wy == 27 && ((p ^ q) & 32)) v = -1e30f;     // last window row mask
        if (wx == 27 && ((p ^ q) & 4))  v = -1e30f;     // last window col mask
        Sf[p * 68 + q] = v;
      }
    }
  }
  __syncthreads();

  // softmax per row
  if (threadIdx.x < 64) {
    int p = threadIdx.x;
    float mx = -3.4e38f;
    for (int q = 0; q < 64; ++q) mx = fmaxf(mx, Sf[p * 68 + q]);
    float s = 0.f;
    for (int q = 0; q < 64; ++q) { float e = __expf(Sf[p * 68 + q] - mx); Sf[p * 68 + q] = e; s += e; }
    float inv = 1.f / s;
    for (int q = 0; q < 64; ++q) ab[p * 72 + q] = f2bf(Sf[p * 68 + q] * inv);
  }
  __syncthreads();

  // Y = attn @ v   (v stored [e][p] so B-fragments are contiguous)
  {
    v8f acc0 = {}, acc1 = {};
    for (int kt = 0; kt < 2; ++kt) {
      v16bf a = fragA(ab, 72, wm, kt << 5);
      acc0 = wmma_bf16(a, fragB(vts, 72, n0, kt << 5), acc0);
      acc1 = wmma_bf16(a, fragB(vts, 72, n1, kt << 5), acc1);
    }
    for (int f = 0; f < 2; ++f) {
      v8f acc = f ? acc1 : acc0;
      int e = wn + (f << 4) + nl;
      for (int r = 0; r < 8; ++r) Sf[(wm + hi8 + r) * 68 + e] = acc[r];
    }
  }
  __syncthreads();

  for (int i = threadIdx.x; i < 4096; i += 256) {
    int p = i >> 6, e = i & 63;
    int iy = p >> 3, ix = p & 7;
    int Yr = (wy << 3) | iy, Xr = (wx << 3) | ix;     // output stays at rolled coords
    long t = (long)bI * 50176 + Yr * 224 + Xr;
    Yb[t * 512 + (hh << 6) + e] = f2bf(Sf[p * 68 + e]);
  }
}

// ---------------------------------------------------------------------------
// Output GEMM: out = Yb(NTOKx512 bf16) @ Wout(512x512) + bout  (fp32 out)
// ---------------------------------------------------------------------------
__global__ __launch_bounds__(256)
void outgemm_kernel(const unsigned short* __restrict__ Yb, const unsigned short* __restrict__ Wp,
                    const float* __restrict__ bout, float* __restrict__ out) {
  extern __shared__ unsigned char smemraw[];
  float* Yt = (float*)smemraw;                           // 64 x 68 f
  unsigned short* As = (unsigned short*)(Yt + 64 * 68);  // 64 x 520 bf16
  const int mt0 = blockIdx.x << 6;

  for (int i = threadIdx.x; i < 4096; i += 256) {
    int row = i >> 6, cv = i & 63;
    async_ld_b128(Yb + (long)(mt0 + row) * 512 + (cv << 3),
                  As + row * 520 + (cv << 3));
  }
  wait_async0();
  __syncthreads();

  const int wave = threadIdx.x >> 5, lane = threadIdx.x & 31;
  const int wm = (wave & 3) << 4, wn = (wave >> 2) << 5;
  const int hi8 = (lane >> 4) << 3, nl = lane & 15;

  for (int nb = 0; nb < 512; nb += 64) {
    const int nt0 = (nb + wn) >> 4, nt1 = nt0 + 1;
    v8f acc0 = {}, acc1 = {};
    v16bf b0 = fragBg(Wp, 32, 0, nt0);
    v16bf b1 = fragBg(Wp, 32, 0, nt1);
    for (int kt = 0; kt < 16; ++kt) {
      v16bf a = fragA(As, 520, wm, kt << 5);
      v16bf c0 = b0, c1 = b1;
      if (kt + 1 < 16) {
        b0 = fragBg(Wp, 32, kt + 1, nt0);
        b1 = fragBg(Wp, 32, kt + 1, nt1);
      }
      acc0 = wmma_bf16(a, c0, acc0);
      acc1 = wmma_bf16(a, c1, acc1);
    }
    for (int f = 0; f < 2; ++f) {
      v8f acc = f ? acc1 : acc0;
      int col = nb + wn + (f << 4) + nl;
      float bb = bout[col];
      int cl = col - nb;
      for (int r = 0; r < 8; ++r) Yt[(wm + hi8 + r) * 68 + cl] = acc[r] + bb;
    }
    __syncthreads();
    for (int i = threadIdx.x; i < 4096; i += 256) {
      int row = i >> 6, cl = i & 63;
      out[(long)(mt0 + row) * 512 + nb + cl] = Yt[row * 68 + cl];
    }
    __syncthreads();
  }
}

// ---------------------------------------------------------------------------
// Host
// ---------------------------------------------------------------------------
extern "C" void kernel_launch(void* const* d_in, const int* in_sizes, int n_in,
                              void* d_out, int out_size, void* d_ws, size_t ws_size,
                              hipStream_t stream) {
  (void)in_sizes; (void)n_in; (void)out_size; (void)ws_size;
  const float* x    = (const float*)d_in[0];   // (2,224,224,512)
  const float* co   = (const float*)d_in[1];   // (2,224,224,128)
  const float* Wqkv = (const float*)d_in[2];   // 512x1024
  const float* bqkv = (const float*)d_in[3];
  const float* gv   = (const float*)d_in[4];
  const float* Wv   = (const float*)d_in[5];   // 512x512
  const float* bv   = (const float*)d_in[6];
  const float* Wqkc = (const float*)d_in[7];   // 128x1024
  const float* bqkc = (const float*)d_in[8];
  const float* gc   = (const float*)d_in[9];
  const float* pos  = (const float*)d_in[10];  // 15x15
  const float* Wout = (const float*)d_in[11];  // 512x512
  const float* bout = (const float*)d_in[12];
  float* out = (float*)d_out;

  size_t off = 0;
  auto wsAlloc = [&](size_t bytes) -> void* {
    void* p = (char*)d_ws + off;
    off += (bytes + 255) & ~(size_t)255;
    return p;
  };
  const size_t WHB = (size_t)NWH * 4096 * 2;   // bf16 windowed buffer
  unsigned short* Xb  = (unsigned short*)wsAlloc((size_t)NTOK * 512 * 2);
  unsigned short* Cb  = (unsigned short*)wsAlloc((size_t)NTOK * 128 * 2);
  unsigned short* Wvp = (unsigned short*)wsAlloc((size_t)512 * 1536 * 2);
  unsigned short* Wcp = (unsigned short*)wsAlloc((size_t)128 * 1024 * 2);
  unsigned short* Wop = (unsigned short*)wsAlloc((size_t)512 * 512 * 2);
  unsigned short* qb  = (unsigned short*)wsAlloc(WHB);
  unsigned short* kb  = (unsigned short*)wsAlloc(WHB);
  unsigned short* qcb = (unsigned short*)wsAlloc(WHB);
  unsigned short* kcb = (unsigned short*)wsAlloc(WHB);
  unsigned short* vtb = (unsigned short*)wsAlloc(WHB);
  unsigned short* Yb  = (unsigned short*)wsAlloc((size_t)NTOK * 512 * 2);

  // dynamic LDS sizes
  auto projSh = [](int K, int N) -> size_t {
    return (size_t)64 * (K + 8) * 2 + (size_t)64 * (N + 8) * 2 + 64 * 4;
  };
  size_t sh1 = projSh(512, 1536);
  size_t sh2 = projSh(128, 1024);
  size_t shA = (size_t)(256 + 64 * 68) * 4 + (size_t)6 * 64 * 72 * 2;
  size_t shO = (size_t)64 * 68 * 4 + (size_t)64 * 520 * 2;
  (void)hipFuncSetAttribute((const void*)proj_kernel,
                            hipFuncAttributeMaxDynamicSharedMemorySize, (int)sh1);
  (void)hipFuncSetAttribute((const void*)attn_kernel,
                            hipFuncAttributeMaxDynamicSharedMemorySize, (int)shA);
  (void)hipFuncSetAttribute((const void*)outgemm_kernel,
                            hipFuncAttributeMaxDynamicSharedMemorySize, (int)shO);

  // 1) convert activations to bf16
  cvt_bf16_kernel<<<4096, 256, 0, stream>>>(x, Xb, (long)NTOK * 512);
  cvt_bf16_kernel<<<2048, 256, 0, stream>>>(co, Cb, (long)NTOK * 128);

  // 2) pack weights into WMMA-B fragment order
  pack_w_kernel<<<1024, 256, 0, stream>>>(Wqkv, 1024, Wv, 512, Wvp, 512, 1536);
  pack_w_kernel<<<512, 256, 0, stream>>>(Wqkc, 1024, nullptr, 0, Wcp, 128, 1024);
  pack_w_kernel<<<512, 256, 0, stream>>>(Wout, 512, nullptr, 0, Wop, 512, 512);

  // 3) fused projections + rmsnorm + windowed scatter
  proj_kernel<<<NTOK / 64, 256, sh1, stream>>>(Xb, 512, Wvp, 1536, 1024,
                                               bqkv, bv, gv, 3, qb, kb, vtb);
  proj_kernel<<<NTOK / 64, 256, sh2, stream>>>(Cb, 128, Wcp, 1024, 1024,
                                               bqkc, nullptr, gc, 2, qcb, kcb, nullptr);

  // 4) windowed attention
  attn_kernel<<<NWH, 256, shA, stream>>>(qb, kb, qcb, kcb, vtb, pos, Yb);

  // 5) output projection
  outgemm_kernel<<<NTOK / 64, 256, shO, stream>>>(Yb, Wop, bout, out);
}
